// LinearAttention_69492570849494
// MI455X (gfx1250) — matrix operand order repaired and checked by reading, not compiled, not run on hardware
//
#include <hip/hip_runtime.h>
#include <stdint.h>

// ---------------- CDNA5 wave32 WMMA types ----------------
typedef __attribute__((ext_vector_type(16))) __bf16 v16bf;
typedef __attribute__((ext_vector_type(2)))  __bf16 v2bf;
typedef __attribute__((ext_vector_type(8)))  float  v8f;

#define DIM   128
#define NTOK  32768
#define NB    8
#define TC    128          // tokens per chunk
#define WT_PITCH 144       // bf16 elems, 288B rows (16B aligned, bank-spread)
#define QK_PITCH 144
#define V_PITCH  136       // 272B rows
#define CW_PITCH 136

struct __align__(16) U128 { uint32_t a, b, c, d; };

union Frag16 {             // 16 bf16 = 8 dwords = one WMMA A/B operand
  v16bf v;
  uint32_t u[8];
  U128 q[2];
};
union Acc8 {               // 16x16 f32 C/D fragment
  v8f v;
  float f[8];
};

// native converts: let clang emit v_cvt_pk_bf16_f32 / v_cvt_bf16_f32
__device__ __forceinline__ uint16_t f2bf(float f) {
  __bf16 h = (__bf16)f;
  uint16_t u; __builtin_memcpy(&u, &h, 2); return u;
}
__device__ __forceinline__ uint32_t pk2(float lo, float hi) {
  v2bf h; h[0] = (__bf16)lo; h[1] = (__bf16)hi;
  uint32_t u; __builtin_memcpy(&u, &h, 4); return u;
}

__device__ __forceinline__ v8f wmma_bf16(const Frag16& a, const Frag16& b, v8f c) {
  return __builtin_amdgcn_wmma_f32_16x16x32_bf16(false, a.v, false, b.v,
                                                 (short)0, c, false, false);
}

// A-fragment (16x32 bf16) built from a global fp32 row-major row.
// lane: row = M, elements e: feature = f0 + (e/8)*16 + (e%8), where
// f0 = kstep*32 + half*8 folded by caller.
__device__ __forceinline__ void load_x_frag(Frag16& a, const float* xrow, int f0) {
  float4 p0 = *(const float4*)(xrow + f0);
  float4 p1 = *(const float4*)(xrow + f0 + 4);
  float4 p2 = *(const float4*)(xrow + f0 + 16);
  float4 p3 = *(const float4*)(xrow + f0 + 20);
  a.u[0] = pk2(p0.x, p0.y); a.u[1] = pk2(p0.z, p0.w);
  a.u[2] = pk2(p1.x, p1.y); a.u[3] = pk2(p1.z, p1.w);
  a.u[4] = pk2(p2.x, p2.y); a.u[5] = pk2(p2.z, p2.w);
  a.u[6] = pk2(p3.x, p3.y); a.u[7] = pk2(p3.z, p3.w);
}

// B-fragment from LDS: two separated 16B runs, K = half*8 + 0..7 and half*8 + 16..23.
__device__ __forceinline__ void load_b32(Frag16& f, const uint16_t* base) {
  f.q[0] = *(const U128*)base;
  f.q[1] = *(const U128*)(base + 16);
}
// A-fragment from LDS: two separated 16B runs (K = +0..7 and K = +16..23).
__device__ __forceinline__ void load_a_split(Frag16& f, const uint16_t* base) {
  f.q[0] = *(const U128*)base;
  f.q[1] = *(const U128*)(base + 16);
}

// Row-softmax of 8 C-fragments spanning 128 features.
// C layout: VGPR r <-> row (token), lanes within a 16-half <-> cols, so the
// reduction is a 16-lane butterfly (masks 8,4,2,1 stay within each half).
__device__ __forceinline__ void softmax8(Acc8* g, float outscale) {
#pragma unroll
  for (int r = 0; r < 8; ++r) {
    float m = g[0].f[r];
#pragma unroll
    for (int t = 1; t < 8; ++t) m = fmaxf(m, g[t].f[r]);
#pragma unroll
    for (int off = 8; off >= 1; off >>= 1) m = fmaxf(m, __shfl_xor(m, off, 32));
    float s = 0.f;
#pragma unroll
    for (int t = 0; t < 8; ++t) { float e = __expf(g[t].f[r] - m); g[t].f[r] = e; s += e; }
#pragma unroll
    for (int off = 8; off >= 1; off >>= 1) s += __shfl_xor(s, off, 32);
    float inv = outscale / s;
#pragma unroll
    for (int t = 0; t < 8; ++t) g[t].f[r] *= inv;
  }
}

// pack one fragment's 8 row values (contiguous tokens) into a 16B chunk
__device__ __forceinline__ U128 pack_rows(const Acc8& a) {
  U128 r;
  r.a = pk2(a.f[0], a.f[1]);
  r.b = pk2(a.f[2], a.f[3]);
  r.c = pk2(a.f[4], a.f[5]);
  r.d = pk2(a.f[6], a.f[7]);
  return r;
}

// ---------------- kernel 0: zero the context accumulator ----------------
__global__ void la_zero_kernel(float* p) {
  p[(size_t)blockIdx.x * 256 + threadIdx.x] = 0.f;
}

// ---------------- kernel 1: context[b] += softmax(q)^T @ softmax(k) ------
__global__ __launch_bounds__(256)
void la_ctx_kernel(const float* __restrict__ x, const float* __restrict__ Wqkv,
                   float* __restrict__ ctx, int chunks_per_block) {
  extern __shared__ char smem[];
  uint16_t* Wt  = (uint16_t*)smem;                          // [256 cols][144]
  uint16_t* qsm = (uint16_t*)(smem + 256 * WT_PITCH * 2);   // [128 feat][144 tok]
  uint16_t* ksm = qsm + 128 * QK_PITCH;                     // [128 feat][144 tok]

  const int tid  = threadIdx.x;
  const int lane = tid & 31, wv = tid >> 5;
  const int l16  = lane & 15, half = lane >> 4;
  const int b    = blockIdx.y;
  const int tok_block0 = blockIdx.x * (chunks_per_block * TC);

  // stage W_q|W_k (cols 0..255) transposed as bf16: Wt[col][row]
  for (int i = tid; i < 128 * 256; i += 256) {
    int r = i >> 8, c = i & 255;
    Wt[c * WT_PITCH + r] = f2bf(Wqkv[r * 384 + c]);
  }
  __syncthreads();

  Acc8 cacc[8];                                  // 16(d) x 128(e) strip, persistent
#pragma unroll
  for (int t = 0; t < 8; ++t)
#pragma unroll
    for (int j = 0; j < 8; ++j) cacc[t].f[j] = 0.f;

#pragma unroll 1
  for (int ch = 0; ch < chunks_per_block; ++ch) {
    const int tokc0 = tok_block0 + ch * TC;      // batch-local chunk start

    // ---- phase 1: qk = x @ W_{q,k}, per-wave 16 tokens ----
    Acc8 acc[16];
#pragma unroll
    for (int t = 0; t < 16; ++t)
#pragma unroll
      for (int j = 0; j < 8; ++j) acc[t].f[j] = 0.f;

    const float* xrow = x + ((size_t)b * NTOK + tokc0 + wv * 16 + l16) * DIM;
#pragma unroll
    for (int kst = 0; kst < 4; ++kst) {
      Frag16 a; load_x_frag(a, xrow, kst * 32 + half * 8);
#pragma unroll
      for (int nt = 0; nt < 16; ++nt) {
        Frag16 bw; load_b32(bw, Wt + (nt * 16 + l16) * WT_PITCH + kst * 32 + half * 8);
        acc[nt].v = wmma_bf16(a, bw, acc[nt].v);
      }
    }
    softmax8(acc,     0.08838834764831845f);     // q * dim^-0.5
    softmax8(acc + 8, 1.0f);                     // k

    // store q,k feature-major (transposed); 8 contiguous tokens per fragment
    // row-group -> one 16B store (base = wv*16 + half*8, 16B aligned).
    const int mtok = wv * 16 + half * 8;
#pragma unroll
    for (int nt = 0; nt < 8; ++nt) {
      const int f = nt * 16 + l16;
      *(U128*)(qsm + f * QK_PITCH + mtok) = pack_rows(acc[nt]);
      *(U128*)(ksm + f * QK_PITCH + mtok) = pack_rows(acc[nt + 8]);
    }
    __syncthreads();

    // ---- phase 2: cacc[d-strip][e] += q^T @ k over 128 tokens ----
#pragma unroll
    for (int kst = 0; kst < 4; ++kst) {
      Frag16 aq;
      load_a_split(aq, qsm + (wv * 16 + l16) * QK_PITCH + kst * 32 + half * 8);
#pragma unroll
      for (int et = 0; et < 8; ++et) {
        Frag16 bk; load_b32(bk, ksm + (et * 16 + l16) * QK_PITCH + kst * 32 + half * 8);
        cacc[et].v = wmma_bf16(aq, bk, cacc[et].v);
      }
    }
    __syncthreads();
  }

  // flush strip with f32 atomics (once per block)
#pragma unroll
  for (int et = 0; et < 8; ++et)
#pragma unroll
    for (int r = 0; r < 8; ++r) {
      const int d = wv * 16 + r + 8 * half;
      const int e = et * 16 + l16;
      atomicAdd(&ctx[((size_t)b * DIM + d) * DIM + e], cacc[et].f[r]);
    }
}

// ---------------- kernel 2: CW^T[b][d'][e] = (ctx^T @ W_out)^T, bf16 ------
__global__ __launch_bounds__(256)
void la_cw_kernel(const float* __restrict__ ctx, const float* __restrict__ Wout,
                  uint16_t* __restrict__ CWt) {
  const int b = blockIdx.x;
  for (int i = threadIdx.x; i < DIM * DIM; i += 256) {
    const int dp = i & 127, e = i >> 7;
    const float* cp = ctx + (size_t)b * DIM * DIM + e;
    float s = 0.f;
#pragma unroll 4
    for (int d = 0; d < DIM; ++d) s += cp[d * DIM] * Wout[d * DIM + dp];
    CWt[(size_t)b * DIM * DIM + dp * DIM + e] = f2bf(s);   // [d'][e]
  }
}

// ---------------- kernel 3: out = softmax(x@W_v) @ CW + b_out ------------
__global__ __launch_bounds__(256)
void la_out_kernel(const float* __restrict__ x, const float* __restrict__ Wqkv,
                   const uint16_t* __restrict__ CWt, const float* __restrict__ bout,
                   float* __restrict__ out, int chunks_per_block) {
  extern __shared__ char smem[];
  uint16_t* Wtv = (uint16_t*)smem;                 // [128 cols][144]
  uint16_t* cwl = Wtv + 128 * WT_PITCH;            // [128 d'][136 e]
  uint16_t* vl  = cwl + 128 * CW_PITCH;            // [128 tok][136 feat]

  const int tid  = threadIdx.x;
  const int lane = tid & 31, wv = tid >> 5;
  const int l16  = lane & 15, half = lane >> 4;
  const int b    = blockIdx.y;
  const int tok_block0 = blockIdx.x * (chunks_per_block * TC);

  // stage W_v (cols 256..383) transposed, and CW^T tile, as bf16
  for (int i = tid; i < 128 * 128; i += 256) {
    int r = i >> 7, c = i & 127;
    Wtv[c * WT_PITCH + r] = f2bf(Wqkv[r * 384 + 256 + c]);
  }
  const uint16_t* cwg = CWt + (size_t)b * DIM * DIM;
  for (int i = tid; i < 128 * 128; i += 256)
    cwl[(i >> 7) * CW_PITCH + (i & 127)] = cwg[i];
  __syncthreads();

  float bo[8];
#pragma unroll
  for (int nt = 0; nt < 8; ++nt) bo[nt] = bout[nt * 16 + l16];

#pragma unroll 1
  for (int ch = 0; ch < chunks_per_block; ++ch) {
    const int tokc0 = tok_block0 + ch * TC;

    // ---- phase A: v = softmax(x @ W_v) for this wave's 16 tokens ----
    Acc8 acc[8];
#pragma unroll
    for (int t = 0; t < 8; ++t)
#pragma unroll
      for (int j = 0; j < 8; ++j) acc[t].f[j] = 0.f;

    const float* xrow = x + ((size_t)b * NTOK + tokc0 + wv * 16 + l16) * DIM;
#pragma unroll
    for (int kst = 0; kst < 4; ++kst) {
      Frag16 a; load_x_frag(a, xrow, kst * 32 + half * 8);
#pragma unroll
      for (int nt = 0; nt < 8; ++nt) {
        Frag16 bw; load_b32(bw, Wtv + (nt * 16 + l16) * WT_PITCH + kst * 32 + half * 8);
        acc[nt].v = wmma_bf16(a, bw, acc[nt].v);
      }
    }
    softmax8(acc, 1.0f);

    // transpose v through LDS, token-major (same-wave region only)
#pragma unroll
    for (int nt = 0; nt < 8; ++nt)
#pragma unroll
      for (int r = 0; r < 8; ++r)
        vl[(wv * 16 + r + 8 * half) * V_PITCH + nt * 16 + l16] = f2bf(acc[nt].f[r]);

    // ---- phase B: out_tile = v @ CW ----
    Acc8 oacc[8];
#pragma unroll
    for (int t = 0; t < 8; ++t)
#pragma unroll
      for (int j = 0; j < 8; ++j) oacc[t].f[j] = 0.f;

#pragma unroll
    for (int kst = 0; kst < 4; ++kst) {
      Frag16 av;
      load_a_split(av, vl + (wv * 16 + l16) * V_PITCH + kst * 32 + half * 8);
#pragma unroll
      for (int nt = 0; nt < 8; ++nt) {
        Frag16 bc; load_b32(bc, cwl + (nt * 16 + l16) * CW_PITCH + kst * 32 + half * 8);
        oacc[nt].v = wmma_bf16(av, bc, oacc[nt].v);
      }
    }

    float* orow = out + ((size_t)b * NTOK + tokc0 + wv * 16) * DIM;
#pragma unroll
    for (int nt = 0; nt < 8; ++nt)
#pragma unroll
      for (int r = 0; r < 8; ++r)
        orow[(r + 8 * half) * DIM + nt * 16 + l16] = oacc[nt].f[r] + bo[nt];
  }
}

// ---------------- host-side launcher ----------------
extern "C" void kernel_launch(void* const* d_in, const int* in_sizes, int n_in,
                              void* d_out, int out_size, void* d_ws, size_t ws_size,
                              hipStream_t stream) {
  (void)in_sizes; (void)n_in; (void)out_size; (void)ws_size;
  const float* x    = (const float*)d_in[0];
  const float* Wqkv = (const float*)d_in[1];
  const float* Wout = (const float*)d_in[2];
  const float* bout = (const float*)d_in[3];
  float* out = (float*)d_out;

  float*    ctx = (float*)d_ws;                                          // 512 KB
  uint16_t* CWt = (uint16_t*)((char*)d_ws + (size_t)NB * DIM * DIM * 4); // 256 KB

  // zero the atomic accumulator (d_ws is poisoned)
  la_zero_kernel<<<dim3(NB * DIM * DIM / 256), 256, 0, stream>>>(ctx);

  // context accumulation: 32 blocks/batch x 1024 tokens (8 chunks of 128)
  const size_t lds1 = 256 * WT_PITCH * 2 + 2 * 128 * QK_PITCH * 2;       // 147456
  la_ctx_kernel<<<dim3(32, NB), 256, lds1, stream>>>(x, Wqkv, ctx, 8);

  // fold context^T @ W_out -> CW^T (bf16)
  la_cw_kernel<<<dim3(NB), 256, 0, stream>>>(ctx, Wout, CWt);

  // output pass: 64 blocks/batch x 512 tokens (4 chunks of 128)
  const size_t lds3 = 128 * WT_PITCH * 2 + 128 * CW_PITCH * 2 + 128 * V_PITCH * 2; // 106496
  la_out_kernel<<<dim3(64, NB), 256, lds3, stream>>>(x, Wqkv, CWt, bout, out, 4);
}
